// SpectrumAdapter_69647189672529
// MI455X (gfx1250) — compile-verified, matrix operand order checked by reading, not run on hardware
//
#include <hip/hip_runtime.h>

// ---------------------------------------------------------------------------
// MI455X (gfx1250, wave32) implementation of SpectrumAdapter reference.
// All large GEMMs through v_wmma_f32_16x16x32_bf16; GEMM staging through the
// Tensor Data Mover (tensor_load_to_lds, TENSORcnt) where layout permits.
// ---------------------------------------------------------------------------

typedef __attribute__((ext_vector_type(16))) __bf16 v16bf;
typedef __attribute__((ext_vector_type(8)))  float  v8f;
typedef unsigned int u32x4 __attribute__((ext_vector_type(4)));
typedef int          i32x4 __attribute__((ext_vector_type(4)));
typedef int          i32x8 __attribute__((ext_vector_type(8)));

#if __has_builtin(__builtin_amdgcn_tensor_load_to_lds) && \
    __has_builtin(__builtin_amdgcn_s_wait_tensorcnt)
#define USE_TDM 1
constexpr bool kUseTdm = true;
#else
constexpr bool kUseTdm = false;
#endif

// ---------------- block-wide reductions (8 waves of 32) --------------------
__device__ __forceinline__ float block_sum(float v, float* sh) {
#pragma unroll
  for (int o = 16; o > 0; o >>= 1) v += __shfl_xor(v, o, 32);
  const int w = threadIdx.x >> 5;
  if ((threadIdx.x & 31) == 0) sh[w] = v;
  __syncthreads();
  float r = 0.f;
#pragma unroll
  for (int i = 0; i < 8; ++i) r += sh[i];
  __syncthreads();
  return r;
}

__device__ __forceinline__ float block_max(float v, float* sh) {
#pragma unroll
  for (int o = 16; o > 0; o >>= 1) v = fmaxf(v, __shfl_xor(v, o, 32));
  const int w = threadIdx.x >> 5;
  if ((threadIdx.x & 31) == 0) sh[w] = v;
  __syncthreads();
  float r = sh[0];
#pragma unroll
  for (int i = 1; i < 8; ++i) r = fmaxf(r, sh[i]);
  __syncthreads();
  return r;
}

// ---------------- fp32 -> bf16 convert -------------------------------------
__global__ __launch_bounds__(256) void cvt_bf16_kernel(
    const float* __restrict__ x, __bf16* __restrict__ y, int n) {
  const int i = blockIdx.x * 256 + threadIdx.x;
  if (i < n) y[i] = (__bf16)x[i];
}

// ---------------- prob prep: diff = q_prob - k_prob, logr = log ratio ------
__global__ __launch_bounds__(256) void prob_prep_kernel(
    const float* __restrict__ q, const float* __restrict__ k,
    __bf16* __restrict__ diff, __bf16* __restrict__ logr) {
  __shared__ float sh[8];
  const size_t row = blockIdx.x;
  const float* qr = q + row * 1024;
  const float* kr = k + row * 1024;
  float tq[4], tk[4], sq = 0.f, sk = 0.f;
#pragma unroll
  for (int i = 0; i < 4; ++i) {
    const int c = threadIdx.x + i * 256;
    tq[i] = tanhf(qr[c]) * 0.499f + 0.5f; sq += tq[i];
    tk[i] = tanhf(kr[c]) * 0.499f + 0.5f; sk += tk[i];
  }
  sq = block_sum(sq, sh);
  sk = block_sum(sk, sh);
  const float iq = 1.f / (sq + 1e-8f);
  const float ik = 1.f / (sk + 1e-8f);
#pragma unroll
  for (int i = 0; i < 4; ++i) {
    const int c = threadIdx.x + i * 256;
    const float qp = fmaxf(tq[i] * iq, 1e-8f);
    const float kp = fmaxf(tk[i] * ik, 1e-8f);
    diff[row * 1024 + c] = (__bf16)(qp - kp);
    logr[row * 1024 + c] = (__bf16)(logf(qp) - logf(kp));
  }
}

// ---------------- row softmax over 2048 cols -> bf16 attn ------------------
__global__ __launch_bounds__(256) void softmax_kernel(
    const float* __restrict__ S, __bf16* __restrict__ attn) {
  __shared__ float sh[8];
  const size_t row = blockIdx.x;
  const float* xr = S + row * 2048;
  float v[8];
  float m = -3.4e38f;
#pragma unroll
  for (int i = 0; i < 8; ++i) { v[i] = xr[threadIdx.x + i * 256]; m = fmaxf(m, v[i]); }
  m = block_max(m, sh);
  float s = 0.f;
#pragma unroll
  for (int i = 0; i < 8; ++i) { v[i] = __expf(v[i] - m); s += v[i]; }
  s = block_sum(s, sh);
  const float inv = 1.f / s;
#pragma unroll
  for (int i = 0; i < 8; ++i)
    attn[row * 2048 + threadIdx.x + i * 256] = (__bf16)(v[i] * inv);
}

// ---------------- row layernorm over 1024 cols -----------------------------
template <int OUT_BF>
__global__ __launch_bounds__(256) void ln_kernel(
    const float* __restrict__ x, float* __restrict__ of, __bf16* __restrict__ ob,
    const float* __restrict__ g, const float* __restrict__ b) {
  __shared__ float sh[8];
  const size_t row = blockIdx.x;
  const float* xr = x + row * 1024;
  float v[4], s = 0.f, s2 = 0.f;
#pragma unroll
  for (int i = 0; i < 4; ++i) {
    v[i] = xr[threadIdx.x + i * 256];
    s += v[i]; s2 += v[i] * v[i];
  }
  s = block_sum(s, sh);
  s2 = block_sum(s2, sh);
  const float mu = s * 0.0009765625f;
  const float var = s2 * 0.0009765625f - mu * mu;
  const float rr = rsqrtf(var + 1e-5f);
#pragma unroll
  for (int i = 0; i < 4; ++i) {
    const int c = threadIdx.x + i * 256;
    const float y = (v[i] - mu) * rr * g[c] + b[c];
    if (OUT_BF) ob[row * 1024 + c] = (__bf16)y;
    else        of[row * 1024 + c] = y;
  }
}

// ---------------------------------------------------------------------------
// TDM helpers: descriptor-driven 2D tile DMA global->LDS.
// Tile = 128 rows x 32 bf16 (64B), LDS padded row stride 80B (40 elems):
//   pad_interval = 16 DWORDs (enc 3), pad_amount = 4 DWORDs (enc 3).
// ---------------------------------------------------------------------------
__device__ __forceinline__ unsigned lds_addr_of(void* p) {
  return (unsigned)(unsigned long long)(__attribute__((address_space(3))) char*)p;
}

#ifdef USE_TDM
__device__ __forceinline__ void tdm_load_tile(const __bf16* gsrc, unsigned lds_byte,
                                              int stride_elems) {
  const unsigned long long ga = (unsigned long long)gsrc;
  u32x4 g0;
  g0[0] = 1u;                                   // count=1 (valid user descriptor)
  g0[1] = lds_byte;                             // lds_addr (bytes)
  g0[2] = (unsigned)ga;                         // global_addr[31:0]
  g0[3] = (unsigned)((ga >> 32) & 0x01ffffffull) | (2u << 30);  // addr[56:32] | type=2
  i32x8 g1;
  g1[0] = (int)((1u << 16) |   // data_size = 2 bytes
                (1u << 20) |   // pad_enable
                (3u << 22) |   // pad_interval: 16 DWORDs
                (3u << 25));   // pad_amount : 4 DWORDs
  g1[1] = (int)0xFFFF0000u;    // tensor_dim0 = 0x7fffffff (lo16 @ bits 63:48)
  g1[2] = (int)0xFFFF7FFFu;    // tensor_dim0 hi16 | tensor_dim1 lo16
  g1[3] = (int)((32u << 16) | 0x7FFFu);  // tensor_dim1 hi16 | tile_dim0 = 32
  g1[4] = 128;                 // tile_dim1 = 128, tile_dim2 = 0
  g1[5] = stride_elems;        // tensor_dim0_stride[31:0] (row stride, elems)
  g1[6] = 0;                   // stride hi16 | dim1_stride lo16
  g1[7] = 0;
  const i32x4 zz4 = {0, 0, 0, 0};
  const i32x8 zz8 = {0, 0, 0, 0, 0, 0, 0, 0};
  // 6-arg clang-23 signature: (v4u g0, v8i g1, v4i g2, v4i g3, v8i g4, i32 cpol)
  __builtin_amdgcn_tensor_load_to_lds(g0, g1, zz4, zz4, zz8, 0);
}
#endif

// ---------------------------------------------------------------------------
// Tiled bf16 WMMA GEMM:  C[M,N] = A[M,K] @ B^T  (B stored [N,K] when BL==0,
// or [K,N] when BL==1 -> transposed while staging into LDS).
// Block tile 128x128, K-step 32, 8 waves; wave tile 64x32 = 8 C tiles (v8f).
// BL==0: double-buffered TDM pipeline (tensor_load_to_lds + s_wait_tensorcnt).
// BL==1: register staging with LDS transpose scatter.
// EPI: 0 = acc*scale -> f32        (sid scores)
//      1 = aux - acc -> bf16       (v - attn_feat)
//      2 = acc + bias + aux -> f32 (f_diff / pre-LN residual)
//      3 = gelu(acc + bias) -> bf16
// ---------------------------------------------------------------------------
constexpr int LSTR = 40;  // padded LDS row stride (bf16 elems); 80B, 16B-aligned

template <int EPI, int BL>
__global__ __launch_bounds__(256) void gemm_bf16(
    const __bf16* __restrict__ A, int lda, long long sA,
    const __bf16* __restrict__ B, int ldb, long long sB,
    float* __restrict__ Cf, __bf16* __restrict__ Cb, int ldc, long long sC,
    const float* __restrict__ aux, int ldaux, long long sAux,
    const float* __restrict__ bias, float scale, int K) {
  constexpr int NBUF = (BL == 0 && kUseTdm) ? 2 : 1;
  __shared__ __align__(16) __bf16 sm[NBUF][2][128 * LSTR];

  const int z = blockIdx.z;
  const __bf16* Ab = A + (size_t)z * sA;
  const __bf16* Bb = B + (size_t)z * sB;
  const int m0 = blockIdx.y * 128;
  const int n0 = blockIdx.x * 128;

  const int tid = threadIdx.x;
  const int wid = tid >> 5;
  const int lane = tid & 31;
  const int wm = (wid & 1) * 64;   // wave M offset (2 waves over M)
  const int wn = (wid >> 1) * 32;  // wave N offset (4 waves over N)
  const int hl = lane >> 4;        // half-wave (K-split in fragments)
  const int lc = lane & 15;        // row/col within 16

  const v8f zero8 = {0.f, 0.f, 0.f, 0.f, 0.f, 0.f, 0.f, 0.f};
  v8f acc[4][2];
#pragma unroll
  for (int i = 0; i < 4; ++i)
#pragma unroll
    for (int j = 0; j < 2; ++j) acc[i][j] = zero8;

  auto compute = [&](const __bf16* bufA, const __bf16* bufB) {
    v16bf af[4], bfr[2];
#pragma unroll
    for (int mt = 0; mt < 4; ++mt) {
      const __bf16* p = bufA + (wm + mt * 16 + lc) * LSTR + hl * 8;
      ((uint4*)&af[mt])[0] = *(const uint4*)p;
      ((uint4*)&af[mt])[1] = *(const uint4*)(p + 16);
    }
#pragma unroll
    for (int nt = 0; nt < 2; ++nt) {
      const __bf16* p = bufB + (wn + nt * 16 + lc) * LSTR + hl * 8;
      ((uint4*)&bfr[nt])[0] = *(const uint4*)p;
      ((uint4*)&bfr[nt])[1] = *(const uint4*)(p + 16);
    }
#pragma unroll
    for (int mt = 0; mt < 4; ++mt)
#pragma unroll
      for (int nt = 0; nt < 2; ++nt)
        acc[mt][nt] = __builtin_amdgcn_wmma_f32_16x16x32_bf16(
            false, af[mt], false, bfr[nt], (short)0, acc[mt][nt], false, false);
  };

  __builtin_prefetch(Ab + (size_t)m0 * lda, 0, 0);   // global_prefetch_b8
  __builtin_prefetch(Bb, 0, 0);
  const int nk = K >> 5;

  if constexpr (BL == 0 && kUseTdm) {
#ifdef USE_TDM
    // ---------------- TDM double-buffered pipeline ----------------
    constexpr int TILE_BYTES = 128 * LSTR * 2;  // 10240 per operand tile
    const unsigned ldsBase = lds_addr_of(&sm[0][0][0]);
    auto tdm_issue = [&](int kb) {
      const unsigned boff = (unsigned)(kb & 1) * (2u * TILE_BYTES);
      tdm_load_tile(Ab + (size_t)m0 * lda + kb * 32, ldsBase + boff, lda);
      tdm_load_tile(Bb + (size_t)n0 * ldb + kb * 32, ldsBase + boff + TILE_BYTES, ldb);
    };
    if (wid == 0) tdm_issue(0);
    for (int kb = 0; kb < nk; ++kb) {
      __builtin_amdgcn_s_wait_tensorcnt(0);  // wave0: DMA done; others: no-op
      __syncthreads();                       // make wave0's LDS writes visible
      if (wid == 0 && kb + 1 < nk) tdm_issue(kb + 1);  // prefetch into other buf
      compute(&sm[kb & 1][0][0], &sm[kb & 1][1][0]);
    }
#endif
  } else {
    // ---------------- register-staged pipeline (handles BL transpose) ------
    __bf16* smA = &sm[0][0][0];
    __bf16* smB = &sm[0][1][0];
    uint4 ra[2], rb[2];
    auto loadAB = [&](int k0) {
#pragma unroll
      for (int i = 0; i < 2; ++i) {
        const int slot = tid + i * 256;
        const int row = slot >> 2, kv = slot & 3;
        ra[i] = *(const uint4*)(Ab + (size_t)(m0 + row) * lda + k0 + kv * 8);
        if (BL == 0) {
          rb[i] = *(const uint4*)(Bb + (size_t)(n0 + row) * ldb + k0 + kv * 8);
        } else {
          const int kr = slot >> 4, nv = slot & 15;
          rb[i] = *(const uint4*)(Bb + (size_t)(k0 + kr) * ldb + n0 + nv * 8);
        }
      }
    };
    auto stage = [&]() {
#pragma unroll
      for (int i = 0; i < 2; ++i) {
        const int slot = tid + i * 256;
        const int row = slot >> 2, kv = slot & 3;
        *(uint4*)(smA + row * LSTR + kv * 8) = ra[i];
        if (BL == 0) {
          *(uint4*)(smB + row * LSTR + kv * 8) = rb[i];
        } else {  // transpose [K,N] -> LDS [N][K]
          const int kr = slot >> 4, nv = slot & 15;
          const __bf16* pv = (const __bf16*)&rb[i];
#pragma unroll
          for (int e = 0; e < 8; ++e) smB[(nv * 8 + e) * LSTR + kr] = pv[e];
        }
      }
    };
    loadAB(0);
    for (int kb = 0; kb < nk; ++kb) {
      __syncthreads();
      stage();
      __syncthreads();
      if (kb + 1 < nk) loadAB((kb + 1) << 5);
      compute(smA, smB);
    }
  }

  // Epilogue: C layout per 16x16 tile -> lane hl/lc holds rows (hl*8..hl*8+7), col lc.
#pragma unroll
  for (int mt = 0; mt < 4; ++mt) {
#pragma unroll
    for (int nt = 0; nt < 2; ++nt) {
      const int col = n0 + wn + nt * 16 + lc;
      const int rb0 = m0 + wm + mt * 16 + hl * 8;
#pragma unroll
      for (int r = 0; r < 8; ++r) {
        const int row = rb0 + r;
        const float va = acc[mt][nt][r];
        const size_t off = (size_t)z * sC + (size_t)row * ldc + col;
        if (EPI == 0) {
          Cf[off] = va * scale;
        } else if (EPI == 1) {
          const float vv = aux[(size_t)z * sAux + (size_t)row * ldaux + col];
          Cb[off] = (__bf16)(vv - va);
        } else if (EPI == 2) {
          const float res = aux[(size_t)z * sAux + (size_t)row * ldaux + col];
          Cf[off] = va + bias[col] + res;
        } else {
          const float xg = va + bias[col];
          Cb[off] = (__bf16)(0.5f * xg * (1.f + erff(xg * 0.70710678118f)));
        }
      }
    }
  }
}

// ---------------------------------------------------------------------------
extern "C" void kernel_launch(void* const* d_in, const int* in_sizes, int n_in,
                              void* d_out, int out_size, void* d_ws, size_t ws_size,
                              hipStream_t stream) {
  (void)in_sizes; (void)n_in; (void)out_size; (void)ws_size;
  const float* q  = (const float*)d_in[0];
  const float* k  = (const float*)d_in[1];
  const float* v  = (const float*)d_in[2];
  const float* Wd = (const float*)d_in[3];
  const float* bd = (const float*)d_in[4];
  const float* lg = (const float*)d_in[5];
  const float* lb = (const float*)d_in[6];
  const float* W1 = (const float*)d_in[7];
  const float* b1 = (const float*)d_in[8];
  const float* W2 = (const float*)d_in[9];
  const float* b2 = (const float*)d_in[10];
  float* out = (float*)d_out;

  // Workspace layout (bytes). ~434 MB total, with two region reuses.
  char* w = (char*)d_ws;
  __bf16* diff = (__bf16*)(w + 0);            // 32 MB  [B,N,D] bf16
  __bf16* logr = (__bf16*)(w + 33554432ll);   // 32 MB
  __bf16* vbf  = (__bf16*)(w + 67108864ll);   // 32 MB
  __bf16* ubf  = (__bf16*)(w + 100663296ll);  // 32 MB  v - attn_feat
  __bf16* hbf  = (__bf16*)(w + 134217728ll);  // 32 MB  layernorm(f_diff)
  __bf16* wdb  = (__bf16*)(w + 167772160ll);  // 2 MB
  __bf16* w1b  = (__bf16*)(w + 169869312ll);  // 8 MB
  __bf16* w2b  = (__bf16*)(w + 178257920ll);  // 8 MB
  float*  fdf  = (float* )(w + 186646528ll);  // 64 MB  f_diff fp32
  float*  Sbuf = (float* )(w + 253755392ll);  // 128 MB scores fp32
  __bf16* act  = (__bf16*)(w + 253755392ll);  //        reuse: gelu act bf16
  __bf16* attn = (__bf16*)(w + 387973120ll);  // 64 MB  attn bf16
  float*  outp = (float* )(w + 387973120ll);  //        reuse: mlp+f_diff fp32

  // 1) bf16 conversions for WMMA operands
  cvt_bf16_kernel<<<dim3(1048576 / 256), 256, 0, stream>>>(Wd, wdb, 1048576);
  cvt_bf16_kernel<<<dim3(4194304 / 256), 256, 0, stream>>>(W1, w1b, 4194304);
  cvt_bf16_kernel<<<dim3(4194304 / 256), 256, 0, stream>>>(W2, w2b, 4194304);
  cvt_bf16_kernel<<<dim3(16777216 / 256), 256, 0, stream>>>(v, vbf, 16777216);

  // 2) prob normalization -> diff / log_ratio (bf16)
  prob_prep_kernel<<<16384, 256, 0, stream>>>(q, k, diff, logr);

  // 3) sid_sim/sqrt(D) = -(diff @ logr^T)/32  [B,2048,2048] fp32
  gemm_bf16<0, 0><<<dim3(16, 16, 8), 256, 0, stream>>>(
      diff, 1024, 2097152ll, logr, 1024, 2097152ll,
      Sbuf, nullptr, 2048, 4194304ll, nullptr, 0, 0ll, nullptr, -0.03125f, 1024);

  // 4) row softmax -> attn bf16
  softmax_kernel<<<16384, 256, 0, stream>>>(Sbuf, attn);

  // 5) u = v - attn @ v   (B operand [K,N] -> LDS transpose path)
  gemm_bf16<1, 1><<<dim3(8, 16, 8), 256, 0, stream>>>(
      attn, 2048, 4194304ll, vbf, 1024, 2097152ll,
      nullptr, ubf, 1024, 2097152ll, v, 1024, 2097152ll, nullptr, 0.f, 2048);

  // 6) f_diff = u @ W_diff^T + b_diff + q  (fp32)
  gemm_bf16<2, 0><<<dim3(8, 128, 1), 256, 0, stream>>>(
      ubf, 1024, 0ll, wdb, 1024, 0ll,
      fdf, nullptr, 1024, 0ll, q, 1024, 0ll, bd, 0.f, 1024);

  // 7) h = layernorm(f_diff) -> bf16
  ln_kernel<1><<<16384, 256, 0, stream>>>(fdf, nullptr, hbf, lg, lb);

  // 8) act = gelu(h @ W1^T + b1) -> bf16 [B*N, 4096]
  gemm_bf16<3, 0><<<dim3(32, 128, 1), 256, 0, stream>>>(
      hbf, 1024, 0ll, w1b, 1024, 0ll,
      nullptr, act, 4096, 0ll, nullptr, 0, 0ll, b1, 0.f, 1024);

  // 9) out_pre = act @ W2^T + b2 + f_diff  (fp32)
  gemm_bf16<2, 0><<<dim3(8, 128, 1), 256, 0, stream>>>(
      act, 4096, 0ll, w2b, 4096, 0ll,
      outp, nullptr, 1024, 0ll, fdf, 1024, 0ll, b2, 0.f, 4096);

  // 10) final layernorm -> d_out fp32
  ln_kernel<0><<<16384, 256, 0, stream>>>(outp, out, nullptr, lg, lb);
}
